// SwinTransformerBlock_24026047054180
// MI455X (gfx1250) — compile-verified, matrix operand order checked by reading, not run on hardware
//
#include <hip/hip_runtime.h>
#include <hip/hip_bf16.h>

// ---------------- problem constants ----------------
#define B_      32
#define HH      56
#define WW_     56
#define CC      192
#define WS_     7
#define SS_     3
#define HEADS_  6
#define HD_     32
#define NN      49
#define NPAD    64
#define NWIN    2048            // B_ * 64 windows
#define MPAD    (NWIN * NPAD)   // 131072 padded windowed rows
#define MTOK    (B_ * HH * WW_) // 100352 tokens
#define HIDDEN_ 768
#define SCALE_  0.17677669529663687f
#define EPS_    1e-5f
#define MLP_CHUNK 12544         // MTOK / 8
#define KSLICE  192             // K staged per LDS tile (64 x 192 f16 = 24 KB)
#define BS_BYTES (64 * KSLICE * 2)

// async global->LDS path availability (guarded; falls back to load+ds_store)
#if defined(__has_builtin)
#if __has_builtin(__builtin_amdgcn_global_load_async_to_lds_b128)
#define USE_ASYNC_LDS 1
#endif
#endif
#ifndef USE_ASYNC_LDS
#define USE_ASYNC_LDS 0
#endif

// ---------------- WMMA fragment types ----------------
typedef __attribute__((ext_vector_type(16))) _Float16 v16h;
typedef __attribute__((ext_vector_type(8)))  _Float16 v8h;
typedef __attribute__((ext_vector_type(8)))  float    v8f;
typedef __attribute__((ext_vector_type(4)))  int      v4i;

#define AS1 __attribute__((address_space(1)))
#define AS3 __attribute__((address_space(3)))

union Frag { v16h v; v8h h[2]; };

// ---------------- workspace layout (bytes) ----------------
static const size_t O_WQ   = 0;                                   // 576x192 f16
static const size_t O_WP   = 221184;                              // 192x192 f16
static const size_t O_W1   = 294912;                              // 768x192 f16
static const size_t O_W2   = 589824;                              // 192x768 f16
static const size_t O_XW   = 1048576;                             // MPAD x 192 f16 (also attn_out, also hmid chunk)
static const size_t O_Q    = O_XW + (size_t)MPAD * CC * 2;        // qkv per-head buffers (also x1 f32 later)
static const size_t O_K    = O_Q  + (size_t)NWIN * HEADS_ * NPAD * HD_ * 2;
static const size_t O_VT   = O_K  + (size_t)NWIN * HEADS_ * NPAD * HD_ * 2;  // (also h2 f16 later)
static const size_t O_ATT  = O_XW;
static const size_t O_X1   = O_Q;
static const size_t O_H2   = O_VT;
static const size_t O_HMID = O_XW;

// ---------------- stage a 64 x KSLICE B-tile into LDS (whole block) ----------------
// Bt: transposed weights [N][ldb], already offset to the K-slice base.
// Stages rows col0..col0+63, KSLICE halves each. Trip count is uniform across
// the block (1536 chunks / 256 threads) so EXEC stays all-ones.
__device__ __forceinline__ void stage_Bt_to_lds(
    const _Float16* __restrict__ Bt, int ldb, int col0,
    _Float16* __restrict__ Bs, int tid) {
  const int chunks = 64 * KSLICE / 8;              // 1536 v8h chunks
  for (int i = tid; i < chunks; i += 256) {
    int row = (i * 8) / KSLICE;
    int off = (i * 8) - row * KSLICE;
    const _Float16* src = Bt + (size_t)(col0 + row) * ldb + off;
    _Float16* dst = Bs + row * KSLICE + off;
#if USE_ASYNC_LDS
    __builtin_amdgcn_global_load_async_to_lds_b128(
        (AS1 v4i*)src, (AS3 v4i*)dst, 0, 0);
#else
    *(v8h*)dst = *(const v8h*)src;                 // global_load_b128 + ds_store_b128
#endif
  }
#if USE_ASYNC_LDS
#if defined(__has_builtin) && __has_builtin(__builtin_amdgcn_s_wait_asynccnt)
  __builtin_amdgcn_s_wait_asynccnt(0);
#else
  asm volatile("s_wait_asynccnt 0" ::: "memory");
#endif
#endif
  __syncthreads();
}

// ---------------- 16x64 wave-tile GEMM over one staged K-slice ----------------
// A fragments stream from global; B fragments come from LDS (ds_load_b128).
__device__ __forceinline__ void gemm_slice_16x64(
    const _Float16* __restrict__ ap,   // A + (row0+l16)*lda + 8*g + ks (per-lane)
    const _Float16* __restrict__ Bs,   // LDS tile 64 x KSLICE
    int l16, int g, v8f acc[4]) {
#pragma unroll
  for (int kk = 0; kk < KSLICE; kk += 32) {
    Frag a;
    a.h[0] = *(const v8h*)(ap + kk);
    a.h[1] = *(const v8h*)(ap + kk + 16);
#pragma unroll
    for (int t = 0; t < 4; ++t) {
      Frag b;
      const _Float16* bp = Bs + (size_t)(t * 16 + l16) * KSLICE + kk + 8 * g;
      b.h[0] = *(const v8h*)(bp);
      b.h[1] = *(const v8h*)(bp + 16);
      acc[t] = __builtin_amdgcn_wmma_f32_16x16x32_f16(
          false, a.v, false, b.v, (short)0, acc[t], false, false);
    }
  }
}

// full GEMM: loop K in KSLICE chunks, staging each chunk in LDS.
// K must be a compile-time constant multiple of KSLICE at the call site.
__device__ __forceinline__ void gemm_16x64_staged(
    const _Float16* __restrict__ A, int lda,
    const _Float16* __restrict__ Bt, int K,
    int row0, int col0, int tid, v8f acc[4], _Float16* Bs) {
  const int lane = tid & 31;
  const int l16 = lane & 15, g = lane >> 4;
  const _Float16* ap = A + (size_t)(row0 + l16) * lda + 8 * g;
#pragma unroll
  for (int ks = 0; ks < 768; ks += KSLICE) {       // bound; real trips = K/KSLICE
    if (ks >= K) break;
    stage_Bt_to_lds(Bt + ks, K, col0, Bs, tid);
    gemm_slice_16x64(ap + ks, Bs, l16, g, acc);
    __syncthreads();                               // before next slice overwrites Bs
  }
}

// ---------------- weight convert + transpose: w (Kd x Nd) f32 -> wT (Nd x Kd) f16 ----------------
__global__ void k_wt(const float* __restrict__ w, _Float16* __restrict__ wT, int Kd, int Nd) {
  int idx = blockIdx.x * 256 + threadIdx.x;
  if (idx < Kd * Nd) {
    int n = idx / Kd, k = idx - n * Kd;
    wT[idx] = (_Float16)w[(size_t)k * Nd + n];
  }
}

// ---------------- LN1 + cyclic shift + window partition (padded to 64 rows/window) ----------------
__global__ __launch_bounds__(256) void k_ln1(
    const float* __restrict__ x, const float* __restrict__ gw, const float* __restrict__ gb,
    _Float16* __restrict__ xw) {
  int lane = threadIdx.x & 31, wave = threadIdx.x >> 5;
  int R = blockIdx.x * 8 + wave;                  // padded windowed row
  int win = R >> 6, n = R & 63;
  _Float16* orow = xw + (size_t)R * CC;
  if (n >= NN) {
#pragma unroll
    for (int i = 0; i < 6; ++i) orow[lane + 32 * i] = (_Float16)0.f;
    return;
  }
  int b = win >> 6, nwi = win & 63;
  int wh = nwi >> 3, wwc = nwi & 7;
  int nh = n / WS_, nwd = n - nh * WS_;
  int hs = (wh * WS_ + nh + SS_) % HH;            // shifted source coords
  int ws = (wwc * WS_ + nwd + SS_) % WW_;
  const float* xr = x + ((size_t)b * (HH * WW_) + hs * WW_ + ws) * CC;
  float v[6]; float s = 0.f;
#pragma unroll
  for (int i = 0; i < 6; ++i) { v[i] = xr[lane + 32 * i]; s += v[i]; }
#pragma unroll
  for (int off = 16; off; off >>= 1) s += __shfl_xor(s, off, 32);
  float mean = s * (1.f / CC);
  float vs = 0.f;
#pragma unroll
  for (int i = 0; i < 6; ++i) { float d = v[i] - mean; vs += d * d; }
#pragma unroll
  for (int off = 16; off; off >>= 1) vs += __shfl_xor(vs, off, 32);
  float inv = rsqrtf(vs * (1.f / CC) + EPS_);
#pragma unroll
  for (int i = 0; i < 6; ++i) {
    int c = lane + 32 * i;
    orow[c] = (_Float16)((v[i] - mean) * inv * gw[c] + gb[c]);
  }
}

// ---------------- QKV GEMM (MPAD x 192) @ (192 x 576) with head scatter ----------------
__global__ __launch_bounds__(256) void k_gemm_qkv(
    const _Float16* __restrict__ A, const _Float16* __restrict__ wT,
    const float* __restrict__ bias,
    _Float16* __restrict__ q, _Float16* __restrict__ k, _Float16* __restrict__ vt) {
  extern __shared__ _Float16 Bs[];
  int row0 = blockIdx.x * 128 + (threadIdx.x >> 5) * 16;
  int col0 = blockIdx.y * 64;
  v8f acc[4] = {};
  gemm_16x64_staged(A, CC, wT, CC, row0, col0, threadIdx.x, acc, Bs);
  int lane = threadIdx.x & 31;
  int l16 = lane & 15, g = lane >> 4;
#pragma unroll
  for (int t = 0; t < 4; ++t)
#pragma unroll
    for (int r = 0; r < 8; ++r) {
      int R = row0 + r + 8 * g;
      int c = col0 + t * 16 + l16;                 // 0..575
      float val = acc[t][r] + bias[c];
      int which = c / CC, cc = c - which * CC;
      int head = cc >> 5, d = cc & 31;
      int win = R >> 6, n = R & 63;
      size_t base = (size_t)(win * HEADS_ + head) * (NPAD * HD_);
      if (which == 0)      q[base + (size_t)n * HD_ + d]  = (_Float16)(val * SCALE_);
      else if (which == 1) k[base + (size_t)n * HD_ + d]  = (_Float16)val;
      else                 vt[base + (size_t)d * NPAD + n] = (_Float16)val;  // transposed [d][key]
    }
}

// ---------------- attention: one (window, head) per 128-thread block ----------------
__global__ __launch_bounds__(128) void k_attn(
    const _Float16* __restrict__ q, const _Float16* __restrict__ k,
    const _Float16* __restrict__ vt, const float* __restrict__ rel,
    const float* __restrict__ mask, _Float16* __restrict__ att) {
  __shared__ __align__(16) float    S[NPAD][NPAD];
  __shared__ __align__(16) _Float16 P[NPAD][NPAD];
  int wh = blockIdx.x;
  int win = wh / HEADS_, head = wh - win * HEADS_;
  int lane = threadIdx.x & 31, wave = threadIdx.x >> 5;
  int l16 = lane & 15, g = lane >> 4;
  const _Float16* qb = q  + (size_t)wh * (NPAD * HD_);
  const _Float16* kb = k  + (size_t)wh * (NPAD * HD_);
  const _Float16* vb = vt + (size_t)wh * (HD_ * NPAD);

  // warm V toward the WGP while S and softmax run (global_prefetch_b8)
  __builtin_prefetch(vb + (size_t)threadIdx.x * 64, 0, 0);

  // ---- S = (q*scale) @ k^T : each wave does a 16x64 strip, K=32 in one WMMA ----
  {
    Frag a;
    const _Float16* ap = qb + (size_t)(wave * 16 + l16) * HD_ + 8 * g;
    a.h[0] = *(const v8h*)(ap);
    a.h[1] = *(const v8h*)(ap + 16);
#pragma unroll
    for (int t = 0; t < 4; ++t) {
      Frag b;
      const _Float16* bp = kb + (size_t)(t * 16 + l16) * HD_ + 8 * g;
      b.h[0] = *(const v8h*)(bp);
      b.h[1] = *(const v8h*)(bp + 16);
      v8f acc = {};
      acc = __builtin_amdgcn_wmma_f32_16x16x32_f16(
          false, a.v, false, b.v, (short)0, acc, false, false);
#pragma unroll
      for (int r = 0; r < 8; ++r)
        S[wave * 16 + r + 8 * g][t * 16 + l16] = acc[r];
    }
  }
  __syncthreads();

  // ---- softmax per query row (+rel-pos bias, +shift mask); padded cols -> 0 ----
  int row = threadIdx.x;
  if (row < NPAD) {
    if (row < NN) {
      int ih = row / WS_, iw = row - ih * WS_;
      const float* mrow = mask + ((size_t)(win & 63) * NN + row) * NN;
      float vmax = -1e30f;
      for (int j = 0; j < NN; ++j) {
        int jh = j / WS_, jw = j - jh * WS_;
        int idx = (ih - jh + WS_ - 1) * (2 * WS_ - 1) + (iw - jw + WS_ - 1);
        float s = S[row][j] + rel[idx * HEADS_ + head] + mrow[j];
        S[row][j] = s;
        vmax = fmaxf(vmax, s);
      }
      float sum = 0.f;
      for (int j = 0; j < NN; ++j) { float e = __expf(S[row][j] - vmax); S[row][j] = e; sum += e; }
      float inv = 1.f / sum;
      for (int j = 0; j < NN; ++j) P[row][j] = (_Float16)(S[row][j] * inv);
      for (int j = NN; j < NPAD; ++j) P[row][j] = (_Float16)0.f;
    } else {
      for (int j = 0; j < NPAD; ++j) P[row][j] = (_Float16)0.f;
    }
  }
  __syncthreads();

  // ---- out = P @ v : 16x32 per wave, K=64 in two WMMA steps ----
  {
    v8f acc[2] = {};
#pragma unroll
    for (int kk = 0; kk < NPAD; kk += 32) {
      Frag a;
      a.h[0] = *(const v8h*)&P[wave * 16 + l16][kk + 8 * g];
      a.h[1] = *(const v8h*)&P[wave * 16 + l16][kk + 16 + 8 * g];
#pragma unroll
      for (int t = 0; t < 2; ++t) {
        Frag b;
        const _Float16* bp = vb + (size_t)(t * 16 + l16) * NPAD + kk + 8 * g;
        b.h[0] = *(const v8h*)(bp);
        b.h[1] = *(const v8h*)(bp + 16);
        acc[t] = __builtin_amdgcn_wmma_f32_16x16x32_f16(
            false, a.v, false, b.v, (short)0, acc[t], false, false);
      }
    }
#pragma unroll
    for (int t = 0; t < 2; ++t)
#pragma unroll
      for (int r = 0; r < 8; ++r) {
        int n = wave * 16 + r + 8 * g;
        int d = t * 16 + l16;
        att[((size_t)win * NPAD + n) * CC + head * HD_ + d] = (_Float16)acc[t][r];
      }
  }
}

// ---------------- proj GEMM + window reverse + unshift + residual -> x1 (f32) ----------------
__global__ __launch_bounds__(256) void k_gemm_proj(
    const _Float16* __restrict__ A, const _Float16* __restrict__ wT,
    const float* __restrict__ bias, const float* __restrict__ x0,
    float* __restrict__ x1) {
  extern __shared__ _Float16 Bs[];
  int row0 = blockIdx.x * 128 + (threadIdx.x >> 5) * 16;
  int col0 = blockIdx.y * 64;
  v8f acc[4] = {};
  gemm_16x64_staged(A, CC, wT, CC, row0, col0, threadIdx.x, acc, Bs);
  int lane = threadIdx.x & 31;
  int l16 = lane & 15, g = lane >> 4;
#pragma unroll
  for (int t = 0; t < 4; ++t)
#pragma unroll
    for (int r = 0; r < 8; ++r) {
      int R = row0 + r + 8 * g;
      int c = col0 + t * 16 + l16;
      int win = R >> 6, n = R & 63;
      if (n < NN) {
        int b = win >> 6, nwi = win & 63;
        int wh = nwi >> 3, wwc = nwi & 7;
        int nh = n / WS_, nwd = n - nh * WS_;
        int hd = (wh * WS_ + nh + SS_) % HH;       // same map as forward read = reverse write
        int wd = (wwc * WS_ + nwd + SS_) % WW_;
        size_t tok = (size_t)b * (HH * WW_) + hd * WW_ + wd;
        x1[tok * CC + c] = x0[tok * CC + c] + acc[t][r] + bias[c];
      }
    }
}

// ---------------- LN2: x1 (f32) -> h2 (f16) ----------------
__global__ __launch_bounds__(256) void k_ln2(
    const float* __restrict__ x1, const float* __restrict__ gw, const float* __restrict__ gb,
    _Float16* __restrict__ h2) {
  int lane = threadIdx.x & 31, wave = threadIdx.x >> 5;
  int R = blockIdx.x * 8 + wave;
  const float* xr = x1 + (size_t)R * CC;
  float v[6]; float s = 0.f;
#pragma unroll
  for (int i = 0; i < 6; ++i) { v[i] = xr[lane + 32 * i]; s += v[i]; }
#pragma unroll
  for (int off = 16; off; off >>= 1) s += __shfl_xor(s, off, 32);
  float mean = s * (1.f / CC);
  float vs = 0.f;
#pragma unroll
  for (int i = 0; i < 6; ++i) { float d = v[i] - mean; vs += d * d; }
#pragma unroll
  for (int off = 16; off; off >>= 1) vs += __shfl_xor(vs, off, 32);
  float inv = rsqrtf(vs * (1.f / CC) + EPS_);
  _Float16* orow = h2 + (size_t)R * CC;
#pragma unroll
  for (int i = 0; i < 6; ++i) {
    int c = lane + 32 * i;
    orow[c] = (_Float16)((v[i] - mean) * inv * gw[c] + gb[c]);
  }
}

// ---------------- fc1 GEMM + bias + exact GELU -> hmid (f16, chunk) ----------------
__global__ __launch_bounds__(256) void k_gemm_fc1(
    const _Float16* __restrict__ A, const _Float16* __restrict__ wT,
    const float* __restrict__ bias, _Float16* __restrict__ hmid) {
  extern __shared__ _Float16 Bs[];
  int row0 = blockIdx.x * 128 + (threadIdx.x >> 5) * 16;
  int col0 = blockIdx.y * 64;
  v8f acc[4] = {};
  gemm_16x64_staged(A, CC, wT, CC, row0, col0, threadIdx.x, acc, Bs);
  int lane = threadIdx.x & 31;
  int l16 = lane & 15, g = lane >> 4;
#pragma unroll
  for (int t = 0; t < 4; ++t)
#pragma unroll
    for (int r = 0; r < 8; ++r) {
      int R = row0 + r + 8 * g;
      int c = col0 + t * 16 + l16;
      float val = acc[t][r] + bias[c];
      float ge = 0.5f * val * (1.f + erff(val * 0.70710678118654752f));
      hmid[(size_t)R * HIDDEN_ + c] = (_Float16)ge;
    }
}

// ---------------- fc2 GEMM + bias + residual -> out (f32, chunk) ----------------
__global__ __launch_bounds__(256) void k_gemm_fc2(
    const _Float16* __restrict__ A, const _Float16* __restrict__ wT,
    const float* __restrict__ bias, const float* __restrict__ x1c,
    float* __restrict__ out) {
  extern __shared__ _Float16 Bs[];
  int row0 = blockIdx.x * 128 + (threadIdx.x >> 5) * 16;
  int col0 = blockIdx.y * 64;
  v8f acc[4] = {};
  gemm_16x64_staged(A, HIDDEN_, wT, HIDDEN_, row0, col0, threadIdx.x, acc, Bs);
  int lane = threadIdx.x & 31;
  int l16 = lane & 15, g = lane >> 4;
#pragma unroll
  for (int t = 0; t < 4; ++t)
#pragma unroll
    for (int r = 0; r < 8; ++r) {
      int R = row0 + r + 8 * g;
      int c = col0 + t * 16 + l16;
      out[(size_t)R * CC + c] = x1c[(size_t)R * CC + c] + acc[t][r] + bias[c];
    }
}

// ---------------- launch ----------------
extern "C" void kernel_launch(void* const* d_in, const int* in_sizes, int n_in,
                              void* d_out, int out_size, void* d_ws, size_t ws_size,
                              hipStream_t stream) {
  const float* x      = (const float*)d_in[0];
  const float* mask   = (const float*)d_in[1];
  const float* rel    = (const float*)d_in[2];
  const float* qkv_w  = (const float*)d_in[3];
  const float* qkv_b  = (const float*)d_in[4];
  const float* proj_w = (const float*)d_in[5];
  const float* proj_b = (const float*)d_in[6];
  const float* n1w    = (const float*)d_in[7];
  const float* n1b    = (const float*)d_in[8];
  const float* n2w    = (const float*)d_in[9];
  const float* n2b    = (const float*)d_in[10];
  const float* fc1_w  = (const float*)d_in[11];
  const float* fc1_b  = (const float*)d_in[12];
  const float* fc2_w  = (const float*)d_in[13];
  const float* fc2_b  = (const float*)d_in[14];

  char* ws = (char*)d_ws;
  _Float16* wqT  = (_Float16*)(ws + O_WQ);
  _Float16* wpT  = (_Float16*)(ws + O_WP);
  _Float16* w1T  = (_Float16*)(ws + O_W1);
  _Float16* w2T  = (_Float16*)(ws + O_W2);
  _Float16* xw   = (_Float16*)(ws + O_XW);
  _Float16* qf   = (_Float16*)(ws + O_Q);
  _Float16* kf   = (_Float16*)(ws + O_K);
  _Float16* vtf  = (_Float16*)(ws + O_VT);
  _Float16* att  = (_Float16*)(ws + O_ATT);
  float*    x1   = (float*)   (ws + O_X1);
  _Float16* h2   = (_Float16*)(ws + O_H2);
  _Float16* hmid = (_Float16*)(ws + O_HMID);

  // weights -> transposed f16
  k_wt<<<(576 * 192 + 255) / 256, 256, 0, stream>>>(qkv_w, wqT, 192, 576);
  k_wt<<<(192 * 192 + 255) / 256, 256, 0, stream>>>(proj_w, wpT, 192, 192);
  k_wt<<<(768 * 192 + 255) / 256, 256, 0, stream>>>(fc1_w, w1T, 192, 768);
  k_wt<<<(192 * 768 + 255) / 256, 256, 0, stream>>>(fc2_w, w2T, 768, 192);

  // LN1 + shift + window partition
  k_ln1<<<MPAD / 8, 256, 0, stream>>>(x, n1w, n1b, xw);

  // QKV
  { dim3 g(MPAD / 128, 576 / 64);
    k_gemm_qkv<<<g, 256, BS_BYTES, stream>>>(xw, wqT, qkv_b, qf, kf, vtf); }

  // attention
  k_attn<<<NWIN * HEADS_, 128, 0, stream>>>(qf, kf, vtf, rel, mask, att);

  // proj + window reverse + residual
  { dim3 g(MPAD / 128, 192 / 64);
    k_gemm_proj<<<g, 256, BS_BYTES, stream>>>(att, wpT, proj_b, x, x1); }

  // LN2
  k_ln2<<<MTOK / 8, 256, 0, stream>>>(x1, n2w, n2b, h2);

  // MLP in 8 row-chunks (bounds hidden buffer to ~19 MB)
  for (int c = 0; c < 8; ++c) {
    size_t ro = (size_t)c * MLP_CHUNK;
    dim3 g1(MLP_CHUNK / 128, HIDDEN_ / 64);
    k_gemm_fc1<<<g1, 256, BS_BYTES, stream>>>(h2 + ro * CC, w1T, fc1_b, hmid);
    dim3 g2(MLP_CHUNK / 128, CC / 64);
    k_gemm_fc2<<<g2, 256, BS_BYTES, stream>>>(hmid, w2T, fc2_b, x1 + ro * CC,
                                              (float*)d_out + ro * CC);
  }
}
